// GGMLLinear_3822520893615
// MI455X (gfx1250) — matrix operand order repaired and checked by reading, not compiled, not run on hardware
//
#include <hip/hip_runtime.h>

typedef __attribute__((ext_vector_type(16))) _Float16 v16h;
typedef __attribute__((ext_vector_type(8)))  float    v8f;
typedef __attribute__((ext_vector_type(2)))  __fp16   fp16x2;

#define NT 256

// pack two fp32 -> dword of two f16 (v_cvt_pk_rtz_f16_f32)
__device__ __forceinline__ unsigned pack2(float lo, float hi) {
  union { fp16x2 h; unsigned u; } c;
  c.h = __builtin_amdgcn_cvt_pkrtz(lo, hi);
  return c.u;
}

// ================= Pass 1a: A fp32 -> f16 =================
__global__ __launch_bounds__(NT)
void cvt_a_f16(const float* __restrict__ A, unsigned short* __restrict__ Ah) {
  const size_t i = ((size_t)blockIdx.x * NT + threadIdx.x) * 8;
  const float4* p = reinterpret_cast<const float4*>(A + i);
  float4 f0 = p[0], f1 = p[1];
  uint4 o = { pack2(f0.x, f0.y), pack2(f0.z, f0.w),
              pack2(f1.x, f1.y), pack2(f1.z, f1.w) };
  *reinterpret_cast<uint4*>(Ah + i) = o;
}

// ================= Pass 1b: Q8 dequant -> f16 =================
// W[o,i] = (q - 128) * s  ==  fma(q, s, -128*s)
__global__ __launch_bounds__(NT)
void dq_w_f16(const int* __restrict__ Wq, const float* __restrict__ Sc,
              unsigned short* __restrict__ Wh) {
  const size_t i = ((size_t)blockIdx.x * NT + threadIdx.x) * 8;  // 8 elems, same Q8 block
  const float s = Sc[i >> 5];
  const float c = -128.0f * s;
  const int4* p = reinterpret_cast<const int4*>(Wq + i);
  int4 q0 = p[0], q1 = p[1];
  uint4 o = { pack2(fmaf((float)q0.x, s, c), fmaf((float)q0.y, s, c)),
              pack2(fmaf((float)q0.z, s, c), fmaf((float)q0.w, s, c)),
              pack2(fmaf((float)q1.x, s, c), fmaf((float)q1.y, s, c)),
              pack2(fmaf((float)q1.z, s, c), fmaf((float)q1.w, s, c)) };
  *reinterpret_cast<uint4*>(Wh + i) = o;
}

// ================= Pass 2: 256x128 f16 GEMM, async global->LDS =================
#define BK2 64
#define BM2 256
#define BN2 128
#define BKP 72   // padded row stride (f16 units): 144 B = 36 banks -> conflict-free frags

// issue 64B (4 x b128) from one half-row; single base, 16B immediate offsets.
// ISA: the imm offset is added to both the LDS and global address -> same imm moves both sides.
__device__ __forceinline__ void async_halfrow(unsigned lds_addr, unsigned long long gaddr) {
  asm volatile("global_load_async_to_lds_b128 %0, %1, off\n\t"
               "global_load_async_to_lds_b128 %0, %1, off offset:16\n\t"
               "global_load_async_to_lds_b128 %0, %1, off offset:32\n\t"
               "global_load_async_to_lds_b128 %0, %1, off offset:48"
               :: "v"(lds_addr), "v"(gaddr) : "memory");
}
__device__ __forceinline__ void wait_async0() {
  asm volatile("s_wait_asynccnt 0x0" ::: "memory");
}

__global__ __launch_bounds__(NT)
void gemm_f16_wmma(const unsigned short* __restrict__ Ah,
                   const unsigned short* __restrict__ Wh,
                   float* __restrict__ C, int M, int N, int K) {
  __shared__ __align__(16) unsigned short As[2][BM2][BKP];   // 2 x 36 KB
  __shared__ __align__(16) unsigned short Bs[2][BN2][BKP];   // 2 x 18 KB

  const int tid   = threadIdx.x;
  const int lane  = tid & 31;
  const int wave  = tid >> 5;
  const int wm    = wave & 3;      // 4 waves along M (64 rows each)
  const int wn    = wave >> 2;     // 2 waves along N (64 cols each)
  const int lmod  = lane & 15;
  const int lhalf = lane >> 4;
  const int m0    = blockIdx.y * BM2;
  const int n0    = blockIdx.x * BN2;

  // ---- async staging plan: half-rows of 64B (4 x 16B chunks) ----
  // A tile: 256 rows x 128B data = 512 half-rows; thread handles h = tid, h = tid+256
  // B tile: 128 rows x 128B data = 256 half-rows; thread handles h = tid
  // (row pad bytes are never written/read)
  unsigned           aL0, aL1, bL0;
  unsigned long long aG0, aG1, bG0;
  {
    const int h0 = tid,        r0 = h0 >> 1, c0 = (h0 & 1) * 32;
    const int h1 = tid + 256,  r1 = h1 >> 1, c1 = (h1 & 1) * 32;
    aL0 = (unsigned)(size_t)&As[0][r0][c0];
    aL1 = (unsigned)(size_t)&As[0][r1][c1];
    bL0 = (unsigned)(size_t)&Bs[0][r0][c0];
    aG0 = (unsigned long long)(size_t)(Ah + (size_t)(m0 + r0) * K + c0);
    aG1 = (unsigned long long)(size_t)(Ah + (size_t)(m0 + r1) * K + c1);
    bG0 = (unsigned long long)(size_t)(Wh + (size_t)(n0 + r0) * K + c0);
  }

  v8f acc[4][4];
#pragma unroll
  for (int i = 0; i < 4; ++i)
#pragma unroll
    for (int j = 0; j < 4; ++j)
      acc[i][j] = (v8f){0.f,0.f,0.f,0.f,0.f,0.f,0.f,0.f};

  // preload k-tile 0 -> buffer 0
  async_halfrow(aL0, aG0);
  async_halfrow(aL1, aG1);
  async_halfrow(bL0, bG0);
  wait_async0();
  __syncthreads();

  const int nk = K / BK2;
  for (int kt = 0; kt < nk; ++kt) {
    const int buf = kt & 1;

    // issue next tile's async loads into the other buffer (overlaps WMMAs)
    if (kt + 1 < nk) {
      const unsigned           aoff = (unsigned)((buf ^ 1) * BM2 * BKP * 2);
      const unsigned           boff = (unsigned)((buf ^ 1) * BN2 * BKP * 2);
      const unsigned long long goff = (unsigned long long)(kt + 1) * (BK2 * 2);
      async_halfrow(aL0 + aoff, aG0 + goff);
      async_halfrow(aL1 + aoff, aG1 + goff);
      async_halfrow(bL0 + boff, bG0 + goff);
    }

    union F { uint4 q[2]; v16h v; };
#pragma unroll
    for (int ks = 0; ks < 2; ++ks) {        // two K=32 WMMA steps per tile
      v16h afrag[4];
#pragma unroll
      for (int tm = 0; tm < 4; ++tm) {
        // A 16x32 f16: lanes 0-15 K 0..7 & 16..23; lanes 16-31 K 8..15 & 24..31
        const unsigned short* p = &As[buf][wm * 64 + tm * 16 + lmod][ks * 32];
        F f;
        f.q[0] = *reinterpret_cast<const uint4*>(p + lhalf * 8);
        f.q[1] = *reinterpret_cast<const uint4*>(p + lhalf * 8 + 16);
        afrag[tm] = f.v;
      }
#pragma unroll
      for (int tn = 0; tn < 4; ++tn) {
        // B 32x16 f16: lane = N column; lanes 0-15 K 0..7 & 16..23; lanes 16-31 K 8..15 & 24..31
        const unsigned short* p = &Bs[buf][wn * 64 + tn * 16 + lmod][ks * 32];
        F f;
        f.q[0] = *reinterpret_cast<const uint4*>(p + lhalf * 8);
        f.q[1] = *reinterpret_cast<const uint4*>(p + lhalf * 8 + 16);
        v16h bfrag = f.v;
#pragma unroll
        for (int tm = 0; tm < 4; ++tm)
          acc[tm][tn] = __builtin_amdgcn_wmma_f32_16x16x32_f16(
              false, afrag[tm], false, bfrag,
              (short)0, acc[tm][tn], false, false);
      }
    }

    wait_async0();
    __syncthreads();
  }

  // epilogue: C 16x16 f32 layout (lane = N, VGPR r -> M = r + 8*(lane>=16))
#pragma unroll
  for (int tm = 0; tm < 4; ++tm) {
#pragma unroll
    for (int tn = 0; tn < 4; ++tn) {
      float* cp = C + (size_t)(m0 + wm * 64 + tm * 16 + lhalf * 8) * N
                    + (n0 + wn * 64 + tn * 16 + lmod);
#pragma unroll
      for (int r = 0; r < 8; ++r)
        cp[(size_t)r * N] = acc[tm][tn][r];
    }
  }
}

// ================= Fallback: fused single-pass =================
__device__ __forceinline__ unsigned dq2(int q0, int q1, float s) {
  return pack2((float)(q0 - 128) * s, (float)(q1 - 128) * s);
}

__global__ __launch_bounds__(NT)
void ggml_q8_gemm_fused(const float* __restrict__ A,
                        const int*   __restrict__ Wq,
                        const float* __restrict__ Sc,
                        float* __restrict__ C,
                        int M, int N, int K) {
  __shared__ __align__(16) unsigned short As[2][128][32];
  __shared__ __align__(16) unsigned short Bs[2][128][32];

  const int tid = threadIdx.x;
  const int lane = tid & 31, wave = tid >> 5;
  const int wm = wave & 3, wn = wave >> 2;
  const int lmod = lane & 15, lhalf = lane >> 4;
  const int m0 = blockIdx.y * 128, n0 = blockIdx.x * 128;
  const int grow = tid >> 1, gcol = (tid & 1) * 16;
  const int nblk = K >> 5;

  const float* aptr = A  + (size_t)(m0 + grow) * K + gcol;
  const int*   wptr = Wq + (size_t)(n0 + grow) * K + gcol;
  const float* sptr = Sc + (size_t)(n0 + grow) * nblk;

  v8f acc[2][4];
#pragma unroll
  for (int i = 0; i < 2; ++i)
#pragma unroll
    for (int j = 0; j < 4; ++j)
      acc[i][j] = (v8f){0.f,0.f,0.f,0.f,0.f,0.f,0.f,0.f};

  {
    const float4* ap = reinterpret_cast<const float4*>(aptr);
    const int4*   wp = reinterpret_cast<const int4*>(wptr);
    float4 a4[4]; int4 w4[4];
#pragma unroll
    for (int j = 0; j < 4; ++j) { a4[j] = ap[j]; w4[j] = wp[j]; }
    const float s = sptr[0];
    uint4 pa0 = { pack2(a4[0].x,a4[0].y), pack2(a4[0].z,a4[0].w),
                  pack2(a4[1].x,a4[1].y), pack2(a4[1].z,a4[1].w) };
    uint4 pa1 = { pack2(a4[2].x,a4[2].y), pack2(a4[2].z,a4[2].w),
                  pack2(a4[3].x,a4[3].y), pack2(a4[3].z,a4[3].w) };
    uint4 pw0 = { dq2(w4[0].x,w4[0].y,s), dq2(w4[0].z,w4[0].w,s),
                  dq2(w4[1].x,w4[1].y,s), dq2(w4[1].z,w4[1].w,s) };
    uint4 pw1 = { dq2(w4[2].x,w4[2].y,s), dq2(w4[2].z,w4[2].w,s),
                  dq2(w4[3].x,w4[3].y,s), dq2(w4[3].z,w4[3].w,s) };
    *reinterpret_cast<uint4*>(&As[0][grow][gcol])     = pa0;
    *reinterpret_cast<uint4*>(&As[0][grow][gcol + 8]) = pa1;
    *reinterpret_cast<uint4*>(&Bs[0][grow][gcol])     = pw0;
    *reinterpret_cast<uint4*>(&Bs[0][grow][gcol + 8]) = pw1;
  }
  __syncthreads();

  const int nk = K >> 5;
  for (int kt = 0; kt < nk; ++kt) {
    const int  buf      = kt & 1;
    const bool has_next = (kt + 1) < nk;

    float4 a4[4]; int4 w4[4]; float s = 0.f;
    if (has_next) {
      const int koff = (kt + 1) << 5;
      const float4* ap = reinterpret_cast<const float4*>(aptr + koff);
      const int4*   wp = reinterpret_cast<const int4*>(wptr + koff);
#pragma unroll
      for (int j = 0; j < 4; ++j) { a4[j] = ap[j]; w4[j] = wp[j]; }
      s = sptr[kt + 1];
    }

    union F { uint4 q[2]; v16h v; };
    v16h afrag[2], bfrag[4];
#pragma unroll
    for (int tm = 0; tm < 2; ++tm) {
      const unsigned short* p = &As[buf][wm * 32 + tm * 16 + lmod][0];
      F f;
      f.q[0] = *reinterpret_cast<const uint4*>(p + lhalf * 8);
      f.q[1] = *reinterpret_cast<const uint4*>(p + lhalf * 8 + 16);
      afrag[tm] = f.v;
    }
#pragma unroll
    for (int tn = 0; tn < 4; ++tn) {
      const unsigned short* p = &Bs[buf][wn * 64 + tn * 16 + lmod][0];
      F f;
      f.q[0] = *reinterpret_cast<const uint4*>(p + lhalf * 8);
      f.q[1] = *reinterpret_cast<const uint4*>(p + lhalf * 8 + 16);
      bfrag[tn] = f.v;
    }
#pragma unroll
    for (int tm = 0; tm < 2; ++tm)
#pragma unroll
      for (int tn = 0; tn < 4; ++tn)
        acc[tm][tn] = __builtin_amdgcn_wmma_f32_16x16x32_f16(
            false, afrag[tm], false, bfrag[tn],
            (short)0, acc[tm][tn], false, false);

    if (has_next) {
      const int nb = buf ^ 1;
      uint4 pa0 = { pack2(a4[0].x,a4[0].y), pack2(a4[0].z,a4[0].w),
                    pack2(a4[1].x,a4[1].y), pack2(a4[1].z,a4[1].w) };
      uint4 pa1 = { pack2(a4[2].x,a4[2].y), pack2(a4[2].z,a4[2].w),
                    pack2(a4[3].x,a4[3].y), pack2(a4[3].z,a4[3].w) };
      uint4 pw0 = { dq2(w4[0].x,w4[0].y,s), dq2(w4[0].z,w4[0].w,s),
                    dq2(w4[1].x,w4[1].y,s), dq2(w4[1].z,w4[1].w,s) };
      uint4 pw1 = { dq2(w4[2].x,w4[2].y,s), dq2(w4[2].z,w4[2].w,s),
                    dq2(w4[3].x,w4[3].y,s), dq2(w4[3].z,w4[3].w,s) };
      *reinterpret_cast<uint4*>(&As[nb][grow][gcol])     = pa0;
      *reinterpret_cast<uint4*>(&As[nb][grow][gcol + 8]) = pa1;
      *reinterpret_cast<uint4*>(&Bs[nb][grow][gcol])     = pw0;
      *reinterpret_cast<uint4*>(&Bs[nb][grow][gcol + 8]) = pw1;
    }
    __syncthreads();
  }

#pragma unroll
  for (int tm = 0; tm < 2; ++tm) {
#pragma unroll
    for (int tn = 0; tn < 4; ++tn) {
      float* cp = C + (size_t)(m0 + wm * 32 + tm * 16 + lhalf * 8) * N
                    + (n0 + wn * 64 + tn * 16 + lmod);
#pragma unroll
      for (int r = 0; r < 8; ++r)
        cp[(size_t)r * N] = acc[tm][tn][r];
    }
  }
}

// ================= host dispatch =================
extern "C" void kernel_launch(void* const* d_in, const int* in_sizes, int n_in,
                              void* d_out, int out_size, void* d_ws, size_t ws_size,
                              hipStream_t stream) {
  const float* A  = (const float*)d_in[0];
  const int*   Wq = (const int*)d_in[1];
  const float* Sc = (const float*)d_in[2];
  float*       C  = (float*)d_out;

  const long wq_n = (long)in_sizes[1];
  const long sc_n = (long)in_sizes[2];
  const int  I    = (int)(32l * wq_n / sc_n);
  const int  O    = (int)(wq_n / I);
  const int  M    = (int)((long)in_sizes[0] / I);

  const size_t needA = (size_t)M * I * sizeof(unsigned short);
  const size_t needW = (size_t)O * I * sizeof(unsigned short);

  if (ws_size >= needA + needW &&
      (M % BM2) == 0 && (O % BN2) == 0 && (I % BK2) == 0 &&
      ((size_t)M * I % (8 * NT)) == 0 && ((size_t)O * I % (8 * NT)) == 0) {
    unsigned short* Ah = (unsigned short*)d_ws;
    unsigned short* Wh = Ah + (size_t)M * I;
    const unsigned gA = (unsigned)((size_t)M * I / (8 * NT));
    const unsigned gW = (unsigned)((size_t)O * I / (8 * NT));
    hipLaunchKernelGGL(cvt_a_f16, dim3(gA), dim3(NT), 0, stream, A, Ah);
    hipLaunchKernelGGL(dq_w_f16,  dim3(gW), dim3(NT), 0, stream, Wq, Sc, Wh);
    dim3 grid(O / BN2, M / BM2);
    hipLaunchKernelGGL(gemm_f16_wmma, grid, dim3(NT), 0, stream, Ah, Wh, C, M, O, I);
  } else {
    dim3 grid(O / 128, M / 128);
    hipLaunchKernelGGL(ggml_q8_gemm_fused, grid, dim3(NT), 0, stream, A, Wq, Sc, C, M, O, I);
  }
}
